// VariationalGCNEncoder_32255204393505
// MI455X (gfx1250) — compile-verified, matrix operand order checked by reading, not run on hardware
//
#include <hip/hip_runtime.h>

typedef __attribute__((ext_vector_type(16))) _Float16 v16h;
typedef __attribute__((ext_vector_type(2)))  _Float16 v2h;
typedef __attribute__((ext_vector_type(8)))  float    v8f;

#define NN 50000
#define NE 800000
#define INCH 256
#define HID 128
#define ZDIM 64

// ---------------------------------------------------------------------------
// Degree / normalization kernels
// ---------------------------------------------------------------------------
__global__ void deg_init_kernel(float* __restrict__ deg, int n) {
  int i = blockIdx.x * blockDim.x + threadIdx.x;
  if (i < n) deg[i] = 1.0f;  // implicit self-loop
}

__global__ void deg_count_kernel(const int* __restrict__ col, float* __restrict__ deg, int e) {
  int i = blockIdx.x * blockDim.x + threadIdx.x;
  if (i < e) atomicAdd(&deg[col[i]], 1.0f);
}

__global__ void deg_rsqrt_kernel(float* __restrict__ deg, int n) {
  int i = blockIdx.x * blockDim.x + threadIdx.x;
  if (i < n) deg[i] = rsqrtf(deg[i]);  // deg >= 1, finite
}

// ---------------------------------------------------------------------------
// Self-loop init:  dst[i][f] = op(src[i][f]) * dis[i]^2  (+ bias[f])
// F fixed at 128, 32 threads per node, 4 floats per thread.
// ---------------------------------------------------------------------------
__global__ void self_init_kernel(const float* __restrict__ src,
                                 const float* __restrict__ dis,
                                 const float* __restrict__ bias,
                                 float* __restrict__ dst,
                                 int n_nodes, int relu_src) {
  int t = blockIdx.x * blockDim.x + threadIdx.x;
  int node = t >> 5;
  int f = (t & 31) * 4;
  if (node >= n_nodes) return;
  float d = dis[node];
  float w = d * d;
  const float4 v = *(const float4*)(src + (size_t)node * HID + f);
  float x0 = v.x, x1 = v.y, x2 = v.z, x3 = v.w;
  if (relu_src) {
    x0 = fmaxf(x0, 0.0f); x1 = fmaxf(x1, 0.0f);
    x2 = fmaxf(x2, 0.0f); x3 = fmaxf(x3, 0.0f);
  }
  float4 o;
  o.x = x0 * w; o.y = x1 * w; o.z = x2 * w; o.w = x3 * w;
  if (bias) {
    o.x += bias[f + 0]; o.y += bias[f + 1];
    o.z += bias[f + 2]; o.w += bias[f + 3];
  }
  *(float4*)(dst + (size_t)node * HID + f) = o;
}

// ---------------------------------------------------------------------------
// Edge scatter:  dst[col[e]][f] += op(src[row[e]][f]) * dis[row]*dis[col]
// 32 threads per edge, 4 floats per thread. Node tensors (25.6 MB) are
// L2-resident on MI455X (192 MB L2) so these random atomics stay on-chip.
// ---------------------------------------------------------------------------
__global__ void edge_scatter_kernel(const float* __restrict__ src,
                                    const float* __restrict__ dis,
                                    const int* __restrict__ row,
                                    const int* __restrict__ col,
                                    float* __restrict__ dst,
                                    int n_edges, int relu_src) {
  int t = blockIdx.x * blockDim.x + threadIdx.x;
  int e = t >> 5;
  int f = (t & 31) * 4;
  if (e >= n_edges) return;
  int r = row[e];
  int c = col[e];
  float norm = dis[r] * dis[c];
  const float4 v = *(const float4*)(src + (size_t)r * HID + f);
  float x0 = v.x, x1 = v.y, x2 = v.z, x3 = v.w;
  if (relu_src) {
    x0 = fmaxf(x0, 0.0f); x1 = fmaxf(x1, 0.0f);
    x2 = fmaxf(x2, 0.0f); x3 = fmaxf(x3, 0.0f);
  }
  float* d = dst + (size_t)c * HID + f;
  atomicAdd(d + 0, x0 * norm);
  atomicAdd(d + 1, x1 * norm);
  atomicAdd(d + 2, x2 * norm);
  atomicAdd(d + 3, x3 * norm);
}

// ---------------------------------------------------------------------------
// WMMA GEMM main loop body, unswitched on FULLM (block-uniform) so the hot
// path has no per-iteration guards around the async DMA.
// ---------------------------------------------------------------------------
template <bool FULLM>
__device__ __forceinline__ void gemm_kloop(const float* __restrict__ A,
                                           const float* __restrict__ B,
                                           int M, int N, int K,
                                           int bm, int bn, int tid,
                                           float (*sAf)[32], _Float16 (*sBt)[32],
                                           v8f (&acc)[2][2]) {
  const int lane = tid & 31;
  const int wave = tid >> 5;
  const int wm = wave >> 1;
  const int wn = wave & 1;
  const int lr = lane & 15;
  const int kb = (lane >> 4) * 8;

  // staging coordinates
  const int ar  = tid >> 1;        // A: row 0..63
  const int acb = (tid & 1) * 16;  // A: col base 0/16
  const int bk  = (tid & 15) * 2;  // B: k-pair base 0,2,..,30
  const int bnb = (tid >> 4) * 8;  // B: n base 0,8,..,56

  const uint32_t ldsA = (uint32_t)(uintptr_t)&sAf[ar][acb];  // LDS byte offset
  const int gm = bm + ar;
  const float* gsrcA = A + (size_t)gm * K + acb;   // loop-carried pointer
  const float* gsrcB = B + (size_t)bk * N + bn + bnb;

  for (int k0 = 0; k0 < K; k0 += 32) {
    // ---- stage A tile: 64 bytes/thread via async DMA to LDS ----
    if (FULLM) {
      asm volatile(
          "global_load_async_to_lds_b128 %0, %1, off\n\t"
          "global_load_async_to_lds_b128 %0, %1, off offset:16\n\t"
          "global_load_async_to_lds_b128 %0, %1, off offset:32\n\t"
          "global_load_async_to_lds_b128 %0, %1, off offset:48"
          :: "v"(ldsA), "v"(gsrcA) : "memory");
    } else {
      float4 f0, f1, f2, f3;
      if (gm < M) {
        const float4* s = (const float4*)gsrcA;
        f0 = s[0]; f1 = s[1]; f2 = s[2]; f3 = s[3];
      } else {
        f0 = f1 = f2 = f3 = make_float4(0.f, 0.f, 0.f, 0.f);
      }
      float4* d = (float4*)&sAf[ar][acb];
      d[0] = f0; d[1] = f1; d[2] = f2; d[3] = f3;
    }
    gsrcA += 32;

    // ---- stage B tile transposed: 2 k-rows x 8 n-cols per thread ----
    {
      float4 a0 = ((const float4*)gsrcB)[0];
      float4 a1 = ((const float4*)gsrcB)[1];
      float4 c0 = ((const float4*)(gsrcB + N))[0];
      float4 c1 = ((const float4*)(gsrcB + N))[1];
      float r0[8] = {a0.x, a0.y, a0.z, a0.w, a1.x, a1.y, a1.z, a1.w};
      float r1[8] = {c0.x, c0.y, c0.z, c0.w, c1.x, c1.y, c1.z, c1.w};
#pragma unroll
      for (int i = 0; i < 8; ++i) {
        v2h p;
        p.x = (_Float16)r0[i];
        p.y = (_Float16)r1[i];
        *(v2h*)&sBt[bnb + i][bk] = p;  // packed ds_store_b32
      }
      gsrcB += (size_t)32 * N;
    }

    if (FULLM) asm volatile("s_wait_asynccnt 0x0" ::: "memory");
    __syncthreads();

    // ---- build fragments (A: fp32 LDS -> f16 regs; B: direct f16) ----
    v16h afrag[2], bfrag[2];
#pragma unroll
    for (int i = 0; i < 2; ++i) {
      const float* pa = &sAf[wm * 32 + i * 16 + lr][kb];
#pragma unroll
      for (int q = 0; q < 8; ++q) {
        afrag[i][q]     = (_Float16)pa[q];
        afrag[i][8 + q] = (_Float16)pa[16 + q];
      }
      const _Float16* pb = &sBt[wn * 32 + i * 16 + lr][kb];
#pragma unroll
      for (int q = 0; q < 8; ++q) {
        bfrag[i][q]     = pb[q];
        bfrag[i][8 + q] = pb[16 + q];
      }
    }
#pragma unroll
    for (int i = 0; i < 2; ++i)
#pragma unroll
      for (int j = 0; j < 2; ++j)
        acc[i][j] = __builtin_amdgcn_wmma_f32_16x16x32_f16(
            false, afrag[i], false, bfrag[j],
            (short)0, acc[i][j], false, false);
    __syncthreads();
  }
}

// ---------------------------------------------------------------------------
// WMMA GEMM:  C[M,N] = A[M,K] @ B[K,N] (+ bias[N])
// fp32 in/out, f16 WMMA operands, f32 accumulate.
// Block tile 64x64, 128 threads = 4 waves (2x2), each wave 2x2 WMMA tiles.
// Requires K % 32 == 0, N % 64 == 0 (true for all three calls).
// ---------------------------------------------------------------------------
__global__ __launch_bounds__(128)
void gemm_wmma_kernel(const float* __restrict__ A,
                      const float* __restrict__ B,
                      const float* __restrict__ bias,
                      float* __restrict__ C,
                      int M, int N, int K) {
  __shared__ float    sAf[64][32];  // [m][k]  fp32 (async DMA target)
  __shared__ _Float16 sBt[64][32];  // [n][k]  f16 (transposed)

  const int tid  = threadIdx.x;
  const int lane = tid & 31;
  const int wave = tid >> 5;
  const int wm = wave >> 1;
  const int wn = wave & 1;
  const int bm = blockIdx.x * 64;
  const int bn = blockIdx.y * 64;

  const int  lr    = lane & 15;
  const int  mlo   = (lane >> 4) * 8;
  const bool fullM = (bm + 64 <= M);  // block-uniform

  v8f acc[2][2];
#pragma unroll
  for (int i = 0; i < 2; ++i)
#pragma unroll
    for (int j = 0; j < 2; ++j)
      acc[i][j] = (v8f){0.f, 0.f, 0.f, 0.f, 0.f, 0.f, 0.f, 0.f};

  if (fullM)
    gemm_kloop<true >(A, B, M, N, K, bm, bn, tid, sAf, sBt, acc);
  else
    gemm_kloop<false>(A, B, M, N, K, bm, bn, tid, sAf, sBt, acc);

  // ---- epilogue: C/D layout -> global, fused bias ----
#pragma unroll
  for (int i = 0; i < 2; ++i) {
#pragma unroll
    for (int j = 0; j < 2; ++j) {
      int n = bn + wn * 32 + j * 16 + lr;
      float bv = bias ? bias[n] : 0.0f;
      int mbase = bm + wm * 32 + i * 16 + mlo;
      if (fullM) {
#pragma unroll
        for (int v = 0; v < 8; ++v)
          C[(size_t)(mbase + v) * N + n] = acc[i][j][v] + bv;
      } else {
#pragma unroll
        for (int v = 0; v < 8; ++v)
          if (mbase + v < M) C[(size_t)(mbase + v) * N + n] = acc[i][j][v] + bv;
      }
    }
  }
}

// ---------------------------------------------------------------------------
// Launch: VGAE encoder.
//   pre1 = Agg(x@W1) + b1
//   agg2 = Agg(relu(pre1))            (relu fused into gathers)
//   mu = agg2@Wmu + bmu ; logstd = agg2@Wls + bls
// (Aggregation commutes with the dense layers, so Agg(h1) is computed once.)
// ---------------------------------------------------------------------------
extern "C" void kernel_launch(void* const* d_in, const int* in_sizes, int n_in,
                              void* d_out, int out_size, void* d_ws, size_t ws_size,
                              hipStream_t stream) {
  const float* x   = (const float*)d_in[0];
  const int*   ei  = (const int*)  d_in[1];
  const float* W1  = (const float*)d_in[2];
  const float* b1  = (const float*)d_in[3];
  const float* Wmu = (const float*)d_in[4];
  const float* bmu = (const float*)d_in[5];
  const float* Wls = (const float*)d_in[6];
  const float* bls = (const float*)d_in[7];

  const int* row = ei;        // edge_index[0]
  const int* col = ei + NE;   // edge_index[1]

  float* ws   = (float*)d_ws;
  float* dis  = ws;                         // NN floats
  float* h    = ws + 51200;                 // NN*HID (x @ W1)
  float* pre1 = h + (size_t)NN * HID;       // NN*HID (pre-activation layer 1)
  float* agg2 = pre1 + (size_t)NN * HID;    // NN*HID (Agg(relu(pre1)))

  float* mu = (float*)d_out;
  float* ls = mu + (size_t)NN * ZDIM;

  // 1. symmetric-normalization coefficients
  deg_init_kernel <<<(NN + 255) / 256, 256, 0, stream>>>(dis, NN);
  deg_count_kernel<<<(NE + 255) / 256, 256, 0, stream>>>(col, dis, NE);
  deg_rsqrt_kernel<<<(NN + 255) / 256, 256, 0, stream>>>(dis, NN);

  // 2. h = x @ W1   (WMMA, M=50000, K=256, N=128)
  gemm_wmma_kernel<<<dim3((NN + 63) / 64, HID / 64), 128, 0, stream>>>(
      x, W1, nullptr, h, NN, HID, INCH);

  // 3. pre1 = Agg(h) + b1
  self_init_kernel<<<((size_t)NN * 32 + 255) / 256, 256, 0, stream>>>(
      h, dis, b1, pre1, NN, 0);
  edge_scatter_kernel<<<((size_t)NE * 32 + 255) / 256, 256, 0, stream>>>(
      h, dis, row, col, pre1, NE, 0);

  // 4. agg2 = Agg(relu(pre1))
  self_init_kernel<<<((size_t)NN * 32 + 255) / 256, 256, 0, stream>>>(
      pre1, dis, nullptr, agg2, NN, 1);
  edge_scatter_kernel<<<((size_t)NE * 32 + 255) / 256, 256, 0, stream>>>(
      pre1, dis, row, col, agg2, NE, 1);

  // 5. heads: mu / logstd  (WMMA, M=50000, K=128, N=64, fused bias)
  gemm_wmma_kernel<<<dim3((NN + 63) / 64, ZDIM / 64), 128, 0, stream>>>(
      agg2, Wmu, bmu, mu, NN, ZDIM, HID);
  gemm_wmma_kernel<<<dim3((NN + 63) / 64, ZDIM / 64), 128, 0, stream>>>(
      agg2, Wls, bls, ls, NN, ZDIM, HID);
}
